// TransformerBlock_20237885899534
// MI455X (gfx1250) — compile-verified, matrix operand order checked by reading, not run on hardware
//
#include <hip/hip_runtime.h>
#include <hip/hip_bf16.h>
#include <stdint.h>

// ---------------------------------------------------------------------------
// Types for CDNA5 WMMA (wave32): 16x16x32 f16 -> f32
// ---------------------------------------------------------------------------
typedef __attribute__((ext_vector_type(16))) _Float16 v16h;
typedef __attribute__((ext_vector_type(8)))  _Float16 v8h;
typedef __attribute__((ext_vector_type(8)))  float    v8f;

__device__ __forceinline__ v16h frag16(const _Float16* p0, const _Float16* p1) {
    union { v16h v; struct { v8h a, b; } s; } u;
    u.s.a = *(const v8h*)p0;   // k-halves 0..7   (8..15 for hi lanes)
    u.s.b = *(const v8h*)p1;   // k-halves 16..23 (24..31 for hi lanes)
    return u.v;
}

__device__ __forceinline__ v8f wmma32(v16h a, v16h b, v8f c) {
    return __builtin_amdgcn_wmma_f32_16x16x32_f16(
        /*neg_a=*/false, a, /*neg_b=*/false, b,
        /*c_mod=*/(short)0, c, /*reuse_a=*/false, /*reuse_b=*/false);
}

// ---------------------------------------------------------------------------
// Problem constants
// ---------------------------------------------------------------------------
#define BB   2
#define SS   2048
#define DD   1024
#define HH   16
#define HD_  64
#define MT   (BB*SS)     // 4096 total rows
#define QKVN (3*DD)      // 3072

// ---------------------------------------------------------------------------
// Weight transpose + f32->f16 convert: src [batch][K][N] f32 -> dst [batch][N][K] f16
// ---------------------------------------------------------------------------
__global__ void transpose_f32_to_f16(const float* __restrict__ src,
                                     _Float16* __restrict__ dst, int K, int N) {
    size_t base = (size_t)blockIdx.z * (size_t)K * (size_t)N;
    size_t idx  = (size_t)blockIdx.x * 256 + threadIdx.x;
    if (idx >= (size_t)K * (size_t)N) return;
    int k = (int)(idx / N), n = (int)(idx % N);
    dst[base + (size_t)n * K + k] = (_Float16)src[base + idx];
}

// ---------------------------------------------------------------------------
// LayerNorm over D=1024, one block per row, f32 in -> f16 out
// ---------------------------------------------------------------------------
__global__ __launch_bounds__(256)
void layernorm_f16(const float* __restrict__ x, const float* __restrict__ g,
                   const float* __restrict__ b, _Float16* __restrict__ out) {
    __shared__ float s1[256], s2[256];
    const int tid = threadIdx.x;
    const size_t row = blockIdx.x;
    const float* xr = x + row * DD;
    float v[4], a0 = 0.f, a1 = 0.f;
    for (int i = 0; i < 4; ++i) {
        v[i] = xr[tid + i * 256];
        a0 += v[i]; a1 += v[i] * v[i];
    }
    s1[tid] = a0; s2[tid] = a1;
    __syncthreads();
    for (int s = 128; s > 0; s >>= 1) {
        if (tid < s) { s1[tid] += s1[tid + s]; s2[tid] += s2[tid + s]; }
        __syncthreads();
    }
    const float mu  = s1[0] * (1.f / DD);
    const float var = s2[0] * (1.f / DD) - mu * mu;
    const float rs  = rsqrtf(var + 1e-5f);
    _Float16* orow = out + row * DD;
    for (int i = 0; i < 4; ++i) {
        int c = tid + i * 256;
        orow[c] = (_Float16)((v[i] - mu) * rs * g[c] + b[c]);
    }
}

// ---------------------------------------------------------------------------
// Tiled WMMA GEMM: out[M][N] = A[M][K](f16,row-major) * Bt[N][K](f16) + bias
// Block tile 128x128, BK=64, 256 threads = 8 waves (4x2), wave tile 32x64.
// Double-buffered LDS (2 x 128x72 halves per matrix, ~72 KB of the 320 KB WGP
// LDS) with global->register->LDS pipelining + global_prefetch two stages out.
// ---------------------------------------------------------------------------
template<bool OUT16, bool GELU, bool RESID>
__global__ __launch_bounds__(256)
void gemm_wmma(const _Float16* __restrict__ A, const _Float16* __restrict__ Bt,
               const float* __restrict__ bias, const float* __restrict__ resid,
               void* __restrict__ out, int M, int N, int K) {
    __shared__ _Float16 As[2][128 * 72];   // 64 halves/row, padded to 72 (144B)
    __shared__ _Float16 Bs[2][128 * 72];
    const int tid  = threadIdx.x;
    const int wave = tid >> 5, lane = tid & 31;
    const int hi   = lane >> 4, l15 = lane & 15;
    const int wm   = wave >> 1;          // 0..3 -> m offset *32
    const int wn   = wave & 1;           // 0..1 -> n offset *64
    const int bm   = blockIdx.y * 128, bn = blockIdx.x * 128;
    const int off  = hi * 8;

    // each thread stages 4 chunks (8 halves) per matrix per BK=64 stage
    const int srow = tid >> 3;           // 0..31 base row (then +32 per chunk)
    const int sch  = (tid & 7) * 8;      // 0..56 halves within the 64-wide row

    v8f acc[2][4] = {};
    v8h ra[4], rb[4];

    const int nk = K >> 6;

    // prologue: stage k-block 0
    for (int i = 0; i < 4; ++i) {
        int row = srow + i * 32;
        ra[i] = *(const v8h*)&A [(size_t)(bm + row) * K + sch];
        rb[i] = *(const v8h*)&Bt[(size_t)(bn + row) * K + sch];
    }
    for (int i = 0; i < 4; ++i) {
        int row = srow + i * 32;
        *(v8h*)&As[0][row * 72 + sch] = ra[i];
        *(v8h*)&Bs[0][row * 72 + sch] = rb[i];
    }
    __syncthreads();

    for (int s = 0; s < nk; ++s) {
        const int cur = s & 1;
        if (s + 2 < nk) {   // warm L2 two stages ahead (global_prefetch_b8)
            int k2 = (s + 2) * 64;
            __builtin_prefetch(&A [(size_t)(bm + srow) * K + k2 + sch], 0, 1);
            __builtin_prefetch(&Bt[(size_t)(bn + srow) * K + k2 + sch], 0, 1);
        }
        if (s + 1 < nk) {   // issue next stage's global loads early
            int k1 = (s + 1) * 64;
            for (int i = 0; i < 4; ++i) {
                int row = srow + i * 32;
                ra[i] = *(const v8h*)&A [(size_t)(bm + row) * K + k1 + sch];
                rb[i] = *(const v8h*)&Bt[(size_t)(bn + row) * K + k1 + sch];
            }
        }
        // compute: 2 k-substeps x (2 m-frags x 4 n-frags) = 16 WMMAs
        for (int kk = 0; kk < 2; ++kk) {
            const int kw = kk * 32 + off;
            for (int mf = 0; mf < 2; ++mf) {
                const _Float16* ap = &As[cur][(wm * 32 + mf * 16 + l15) * 72 + kw];
                v16h af = frag16(ap, ap + 16);
                for (int nf = 0; nf < 4; ++nf) {
                    const _Float16* bp = &Bs[cur][(wn * 64 + nf * 16 + l15) * 72 + kw];
                    v16h bf = frag16(bp, bp + 16);
                    acc[mf][nf] = wmma32(af, bf, acc[mf][nf]);
                }
            }
        }
        if (s + 1 < nk) {   // fill the other buffer while peers may still read cur
            for (int i = 0; i < 4; ++i) {
                int row = srow + i * 32;
                *(v8h*)&As[cur ^ 1][row * 72 + sch] = ra[i];
                *(v8h*)&Bs[cur ^ 1][row * 72 + sch] = rb[i];
            }
        }
        __syncthreads();
    }

    for (int mf = 0; mf < 2; ++mf)
        for (int nf = 0; nf < 4; ++nf)
            for (int i = 0; i < 8; ++i) {
                int row = bm + wm * 32 + mf * 16 + i + hi * 8;
                int col = bn + wn * 64 + nf * 16 + l15;
                float v = acc[mf][nf][i] + bias[col];
                if (GELU)  v = 0.5f * v * (1.f + erff(v * 0.70710678118f));
                if (RESID) v += resid[(size_t)row * N + col];
                if (OUT16) ((_Float16*)out)[(size_t)row * N + col] = (_Float16)v;
                else       ((float*)out)   [(size_t)row * N + col] = v;
            }
}

// ---------------------------------------------------------------------------
// Causal flash attention, one block = (64 q-rows, one (b,h)); 4 waves x 16 rows.
// QKV buffer: [4096][3072] f16 (cols 0:1024 Q, 1024:2048 K, 2048:3072 V,
// each section head-major h*64+hd). CTX out: [4096][1024] f16.
// ---------------------------------------------------------------------------
__global__ __launch_bounds__(128)
void flash_attn(const _Float16* __restrict__ QKV, _Float16* __restrict__ CTX) {
    __shared__ _Float16 Kt[64 * 72];       // [j][d]: row-major == B^T layout for QK^T
    __shared__ _Float16 Vt[64 * 72];       // [d][j]: transposed V == B^T layout for P*V
    __shared__ _Float16 Psm[4 * 16 * 64];  // per-wave P tile staging
    const int tid  = threadIdx.x;
    const int wave = tid >> 5, lane = tid & 31;
    const int hi   = lane >> 4, l15 = lane & 15, off = hi * 8;
    const int qtile = blockIdx.x;          // 0..31
    const int bh    = blockIdx.y;          // b*H + h
    const int b = bh / HH, h = bh % HH;
    const size_t rowbase = (size_t)b * SS;

    // Q fragments (A-layout) straight from global, once.
    const int qrow = qtile * 64 + wave * 16 + l15;
    const _Float16* qp = QKV + (rowbase + qrow) * QKVN + h * HD_;
    v16h qf0 = frag16(qp +      off, qp +      off + 16);
    v16h qf1 = frag16(qp + 32 + off, qp + 32 + off + 16);

    v8f   ctx[4] = {};
    float mrow[8], lrow[8];
    for (int i = 0; i < 8; ++i) { mrow[i] = -__builtin_inff(); lrow[i] = 0.f; }
    _Float16* Pw = Psm + wave * (16 * 64);

    for (int t = 0; t <= qtile; ++t) {
        // prefetch next KV tile into L2 while we work on this one
        if (t + 1 <= qtile) {
            int jn = (tid >> 1) & 63, half = (tid & 1);
            __builtin_prefetch(
                &QKV[(rowbase + (t + 1) * 64 + jn) * QKVN + (1 + half) * DD + h * HD_],
                0, 1);
        }
        // Stage K tile row-major (vector copies) and V tile transposed (scalar).
        for (int c = tid; c < 512; c += 128) {          // 64 rows x 8 chunks
            int j = c >> 3, ch = (c & 7) * 8;
            *(v8h*)&Kt[j * 72 + ch] =
                *(const v8h*)&QKV[(rowbase + t * 64 + j) * QKVN + DD + h * HD_ + ch];
        }
        for (int c = tid; c < 4096; c += 128) {
            int j = c >> 6, d = c & 63;
            Vt[d * 72 + j] = QKV[(rowbase + t * 64 + j) * QKVN + 2 * DD + h * HD_ + d];
        }
        __syncthreads();

        // ---- scores S = Q K^T (16x64 per wave: 4 n-tiles x 2 k-steps) ----
        v8f sc[4];
        for (int nf = 0; nf < 4; ++nf) {
            v8f s = {};
            const _Float16* bp = &Kt[(nf * 16 + l15) * 72 + off];
            s = wmma32(qf0, frag16(bp,      bp + 16),      s);
            s = wmma32(qf1, frag16(bp + 32, bp + 32 + 16), s);
            sc[nf] = s;
        }

        // ---- scale, causal mask, row max ----
        const bool diag = (t == qtile);
        float rmax[8];
        for (int i = 0; i < 8; ++i) rmax[i] = -__builtin_inff();
        for (int nf = 0; nf < 4; ++nf)
            for (int i = 0; i < 8; ++i) {
                float v = sc[nf][i] * 0.125f;   // 1/sqrt(64)
                int r = qtile * 64 + wave * 16 + i + hi * 8;
                int c = t * 64 + nf * 16 + l15;
                if (diag && c > r) v = -__builtin_inff();
                sc[nf][i] = v;
                rmax[i] = fmaxf(rmax[i], v);
            }
        for (int m = 1; m < 16; m <<= 1)
            for (int i = 0; i < 8; ++i)
                rmax[i] = fmaxf(rmax[i], __shfl_xor(rmax[i], m));

        // ---- online softmax update ----
        float corr[8], rsum[8];
        for (int i = 0; i < 8; ++i) {
            float mn = fmaxf(mrow[i], rmax[i]);
            corr[i] = __expf(mrow[i] - mn);
            mrow[i] = mn;
            rsum[i] = 0.f;
        }
        for (int nf = 0; nf < 4; ++nf)
            for (int i = 0; i < 8; ++i) {
                float p = __expf(sc[nf][i] - mrow[i]);
                rsum[i] += p;
                Pw[(i + hi * 8) * 64 + nf * 16 + l15] = (_Float16)p;
            }
        for (int m = 1; m < 16; m <<= 1)
            for (int i = 0; i < 8; ++i)
                rsum[i] += __shfl_xor(rsum[i], m);
        for (int i = 0; i < 8; ++i) lrow[i] = lrow[i] * corr[i] + rsum[i];
        for (int df = 0; df < 4; ++df)
            for (int i = 0; i < 8; ++i) ctx[df][i] *= corr[i];

        // ---- ctx += P V  (P re-read from LDS as A-fragments) ----
        for (int ks = 0; ks < 2; ++ks) {
            const _Float16* pp = &Pw[l15 * 64 + ks * 32 + off];
            v16h pf = frag16(pp, pp + 16);
            for (int df = 0; df < 4; ++df) {
                const _Float16* vp = &Vt[(df * 16 + l15) * 72 + ks * 32 + off];
                ctx[df] = wmma32(pf, frag16(vp, vp + 16), ctx[df]);
            }
        }
        __syncthreads();
    }

    // normalize + write concat-head context (f16)
    for (int df = 0; df < 4; ++df)
        for (int i = 0; i < 8; ++i) {
            int r = qtile * 64 + wave * 16 + i + hi * 8;
            int c = h * HD_ + df * 16 + l15;
            CTX[(rowbase + r) * DD + c] = (_Float16)(ctx[df][i] / lrow[i]);
        }
}

// ---------------------------------------------------------------------------
// Host-side orchestration
// ---------------------------------------------------------------------------
extern "C" void kernel_launch(void* const* d_in, const int* in_sizes, int n_in,
                              void* d_out, int out_size, void* d_ws, size_t ws_size,
                              hipStream_t stream) {
    (void)in_sizes; (void)n_in; (void)out_size; (void)ws_size;
    const float* x  = (const float*)d_in[0];
    const float* Wq = (const float*)d_in[1];
    const float* bq = (const float*)d_in[2];
    const float* Wk = (const float*)d_in[3];
    const float* bk = (const float*)d_in[4];
    const float* Wv = (const float*)d_in[5];
    const float* bv = (const float*)d_in[6];
    const float* Wo = (const float*)d_in[7];
    const float* bo = (const float*)d_in[8];
    const float* g1 = (const float*)d_in[9];
    const float* b1 = (const float*)d_in[10];
    const float* g2 = (const float*)d_in[11];
    const float* b2 = (const float*)d_in[12];
    const float* W1 = (const float*)d_in[13];
    const float* c1 = (const float*)d_in[14];
    const float* W2 = (const float*)d_in[15];
    const float* c2 = (const float*)d_in[16];

    uint8_t* ws = (uint8_t*)d_ws;
    const size_t MiB = 1u << 20;
    _Float16* hA    = (_Float16*)(ws + 0 * MiB);    // [4096][1024]   8 MiB
    _Float16* WqkvT = (_Float16*)(ws + 8 * MiB);    // [3072][1024]   6 MiB
    _Float16* WoT   = (_Float16*)(ws + 14 * MiB);   // [1024][1024]   2 MiB
    _Float16* W1T   = (_Float16*)(ws + 16 * MiB);   // [4096][1024]   8 MiB
    _Float16* W2T   = (_Float16*)(ws + 24 * MiB);   // [1024][4096]   8 MiB
    _Float16* QKV   = (_Float16*)(ws + 32 * MiB);   // [4096][3072]  24 MiB
    _Float16* CTX   = (_Float16*)(ws + 56 * MiB);   // [4096][1024]   8 MiB
    float*    X1    = (float*)   (ws + 64 * MiB);   // [4096][1024]  16 MiB
    _Float16* h2    = (_Float16*)(ws + 80 * MiB);   // [4096][1024]   8 MiB
    _Float16* MID   = (_Float16*)(ws + 88 * MiB);   // [4096][4096]  32 MiB
    float*    bias3 = (float*)   (ws + 120 * MiB);  // [3072]

    // ---- 1. weight transpose/convert to f16 [N][K] ----
    transpose_f32_to_f16<<<dim3((DD * HD_ + 255) / 256, 1, HH), 256, 0, stream>>>(
        Wq, WqkvT, DD, HD_);
    transpose_f32_to_f16<<<dim3((DD * HD_ + 255) / 256, 1, HH), 256, 0, stream>>>(
        Wk, WqkvT + (size_t)DD * DD, DD, HD_);
    transpose_f32_to_f16<<<dim3((DD * HD_ + 255) / 256, 1, HH), 256, 0, stream>>>(
        Wv, WqkvT + (size_t)2 * DD * DD, DD, HD_);
    transpose_f32_to_f16<<<dim3((DD * DD + 255) / 256, 1, 1), 256, 0, stream>>>(
        Wo, WoT, DD, DD);
    transpose_f32_to_f16<<<dim3((DD * 4 * DD + 255) / 256, 1, 1), 256, 0, stream>>>(
        W1, W1T, DD, 4 * DD);
    transpose_f32_to_f16<<<dim3((4 * DD * DD + 255) / 256, 1, 1), 256, 0, stream>>>(
        W2, W2T, 4 * DD, DD);

    // ---- 2. pack qkv biases into one [3072] vector ----
    hipMemcpyAsync(bias3,          bq, DD * sizeof(float), hipMemcpyDeviceToDevice, stream);
    hipMemcpyAsync(bias3 + DD,     bk, DD * sizeof(float), hipMemcpyDeviceToDevice, stream);
    hipMemcpyAsync(bias3 + 2 * DD, bv, DD * sizeof(float), hipMemcpyDeviceToDevice, stream);

    // ---- 3. LN1 -> f16 activations ----
    layernorm_f16<<<MT, 256, 0, stream>>>(x, g1, b1, hA);

    // ---- 4. fused QKV projection: [4096][1024] x [1024][3072] ----
    gemm_wmma<true, false, false><<<dim3(QKVN / 128, MT / 128), 256, 0, stream>>>(
        hA, WqkvT, bias3, nullptr, QKV, MT, QKVN, DD);

    // ---- 5. causal flash attention ----
    flash_attn<<<dim3(SS / 64, BB * HH), 128, 0, stream>>>(QKV, CTX);

    // ---- 6. output projection + residual: X1 = x + CTX @ Wo + bo ----
    gemm_wmma<false, false, true><<<dim3(DD / 128, MT / 128), 256, 0, stream>>>(
        CTX, WoT, bo, x, X1, MT, DD, DD);

    // ---- 7. LN2 -> f16 ----
    layernorm_f16<<<MT, 256, 0, stream>>>(X1, g2, b2, h2);

    // ---- 8. MLP up + exact GELU: [4096][1024] x [1024][4096] ----
    gemm_wmma<true, true, false><<<dim3(4 * DD / 128, MT / 128), 256, 0, stream>>>(
        h2, W1T, c1, nullptr, MID, MT, 4 * DD, DD);

    // ---- 9. MLP down + residual -> d_out (fp32) ----
    gemm_wmma<false, false, true><<<dim3(DD / 128, MT / 128), 256, 0, stream>>>(
        MID, W2T, c2, X1, d_out, MT, DD, 4 * DD);
}
